// Sequence_45990509805770
// MI455X (gfx1250) — compile-verified, matrix operand order checked by reading, not run on hardware
//
#include <hip/hip_runtime.h>

// ----------------------------------------------------------------------------
// 2-layer LSTM scan (B=2048, S=512, H1=32, H2=64) for MI455X / gfx1250.
//
// Roofline: ~60 GFLOP total, ~8 MB HBM traffic -> at 23.3 TB/s memory is
// negligible; the kernel is latency-bound on the S=512 serial recurrence.
// Strategy: one wave32 per 16-row batch tile runs the whole scan with all
// weights resident in LDS as pre-swizzled f16 WMMA B-fragments; matmuls use
// v_wmma_f32_16x16x32_f16 (K=32 == H1, H2 = 2 K-passes). 56 WMMAs/step/wave.
// ----------------------------------------------------------------------------

typedef __attribute__((ext_vector_type(16))) _Float16 v16h;
typedef __attribute__((ext_vector_type(8)))  float    v8f;

#define H1 32
#define H2 64

__device__ __forceinline__ float fast_sigmoid(float x) {
  float e = __builtin_amdgcn_exp2f(-1.44269504089f * x);   // v_exp_f32
  return __builtin_amdgcn_rcpf(1.0f + e);                  // v_rcp_f32
}
__device__ __forceinline__ float fast_tanh(float x) {
  // tanh(x) = 2*sigmoid(2x) - 1
  float e = __builtin_amdgcn_exp2f(-2.88539008178f * x);
  return 2.0f * __builtin_amdgcn_rcpf(1.0f + e) - 1.0f;
}
__device__ __forceinline__ void lds_fence() {
  asm volatile("s_wait_dscnt 0" ::: "memory");             // CDNA5 split counter
}

// K index for element e (0..15) of a 16-bit A/B fragment, half-wave hl (0/1).
// Per ISA 7.12.2: VGPR v<4 -> K base 0, v>=4 -> K base 16; lanes 16-31 add 8.
__device__ __forceinline__ int frag_k(int e, int hl) {
  return ((e >> 3) << 4) + (hl << 3) + (e & 7);
}

// Load an A fragment (16 x K, f16) from an LDS row-major stage buffer.
__device__ __forceinline__ v16h load_A(const _Float16* stage, int K, int m, int hl) {
  v16h a;
#pragma unroll
  for (int e = 0; e < 16; ++e) a[e] = stage[m * K + frag_k(e, hl)];
  return a;
}

// Load a pre-swizzled B fragment: fragbuf holds [frag][lane][16 halves].
__device__ __forceinline__ v16h load_B(const _Float16* fragbuf, int fragIdx, int lane) {
  return *(const v16h*)(fragbuf + (fragIdx << 9) + (lane << 4));
}

__device__ __forceinline__ v8f wmma_f16(v16h a, v16h b, v8f c) {
  return __builtin_amdgcn_wmma_f32_16x16x32_f16(false, a, false, b, (short)0, c,
                                                false, false);
}

__global__ void __launch_bounds__(32)
lstm_scan_kernel(const float* __restrict__ inputs,
                 const float* __restrict__ h1_in, const float* __restrict__ c1_in,
                 const float* __restrict__ h2_in, const float* __restrict__ c2_in,
                 const float* __restrict__ W_ih1, const float* __restrict__ W_hh1,
                 const float* __restrict__ b_ih1, const float* __restrict__ b_hh1,
                 const float* __restrict__ W_ih2, const float* __restrict__ W_hh2,
                 const float* __restrict__ b_ih2, const float* __restrict__ b_hh2,
                 const float* __restrict__ W_out, const float* __restrict__ b_out,
                 float* __restrict__ out, int S)
{
  // ---- LDS: weights as B-fragments (f16) + state staging ---- (62,784 B)
  __shared__ __align__(32) _Float16 sWhh1[8  * 512];  // (128,32)  -> 8 frags
  __shared__ __align__(32) _Float16 sWih2[16 * 512];  // (256,32)  -> 16 frags
  __shared__ __align__(32) _Float16 sWhh2[32 * 512];  // (256,64)  -> 16x2 frags
  __shared__ __align__(32) _Float16 sH1[16 * H1];
  __shared__ __align__(32) _Float16 sH2[16 * H2];
  __shared__ float sB1[4 * H1];   // b_ih1 + b_hh1
  __shared__ float sB2[4 * H2];   // b_ih2 + b_hh2
  __shared__ float sWi1[4 * H1];  // W_ih1[:,0]
  __shared__ float sWo[H2];       // W_out[0,:]
  __shared__ float sX[16];

  const int lane  = threadIdx.x;   // one wave per block
  const int nl    = lane & 15;     // column within 16-wide tile
  const int hl    = lane >> 4;     // half-wave
  const int bbase = blockIdx.x * 16;

  // ---- one-time init: biases, scalars, fragment swizzle, state staging ----
  for (int i = lane; i < 4 * H1; i += 32) { sB1[i] = b_ih1[i] + b_hh1[i]; sWi1[i] = W_ih1[i]; }
  for (int i = lane; i < 4 * H2; i += 32) sB2[i] = b_ih2[i] + b_hh2[i];
  for (int i = lane; i < H2;     i += 32) sWo[i] = W_out[i];

  // B[k][n] = W[n][k]; fragment element (lane ln, elem e) covers (n, k).
  for (int f = lane; f < 8 * 512; f += 32) {
    int nt = f >> 9, r = f & 511, ln = r >> 4, e = r & 15;
    int n = nt * 16 + (ln & 15), k = frag_k(e, ln >> 4);
    sWhh1[f] = (_Float16)W_hh1[n * H1 + k];
  }
  for (int f = lane; f < 16 * 512; f += 32) {
    int nt = f >> 9, r = f & 511, ln = r >> 4, e = r & 15;
    int n = nt * 16 + (ln & 15), k = frag_k(e, ln >> 4);
    sWih2[f] = (_Float16)W_ih2[n * H1 + k];
  }
  for (int f = lane; f < 32 * 512; f += 32) {
    int fi = f >> 9, r = f & 511, ln = r >> 4, e = r & 15;
    int nt = fi >> 1, ks = fi & 1;
    int n = nt * 16 + (ln & 15), k = ks * 32 + frag_k(e, ln >> 4);
    sWhh2[f] = (_Float16)W_hh2[n * H2 + k];
  }
  for (int i = lane; i < 16 * H1; i += 32)
    sH1[i] = (_Float16)h1_in[(bbase + (i >> 5)) * H1 + (i & 31)];
  for (int i = lane; i < 16 * H2; i += 32)
    sH2[i] = (_Float16)h2_in[(bbase + (i >> 6)) * H2 + (i & 63)];
  __syncthreads();
  lds_fence();

  const float bo = b_out[0];

  // Persistent fp32 cell state in C-layout: row m = v + 8*hl, col = ht*16 + nl.
  float c1r[2][8], c2r[4][8];
#pragma unroll
  for (int ht = 0; ht < 2; ++ht)
#pragma unroll
    for (int v = 0; v < 8; ++v)
      c1r[ht][v] = c1_in[(bbase + v + 8 * hl) * H1 + ht * 16 + nl];
#pragma unroll
  for (int ht = 0; ht < 4; ++ht)
#pragma unroll
    for (int v = 0; v < 8; ++v)
      c2r[ht][v] = c2_in[(bbase + v + 8 * hl) * H2 + ht * 16 + nl];

  v16h h1A = load_A(sH1, H1, nl, hl);  // h1_{t-1}

  for (int t = 0; t < S; ++t) {
    // stage x_t for this wave's 16 batch rows
    if (lane < 16) sX[lane] = inputs[(bbase + lane) * S + t];
    lds_fence();
    float xm[8];
#pragma unroll
    for (int v = 0; v < 8; ++v) xm[v] = sX[v + 8 * hl];

    // ---------------- layer 1: gates = x*w_ih1 + b + h1 @ W_hh1^T ----------
#pragma unroll
    for (int ht = 0; ht < 2; ++ht) {
      const int nti = ht, ntf = 2 + ht, ntg = 4 + ht, nto = 6 + ht;
      float bi = sB1[nti * 16 + nl], bf = sB1[ntf * 16 + nl];
      float bg = sB1[ntg * 16 + nl], b4 = sB1[nto * 16 + nl];
      float wi = sWi1[nti * 16 + nl], wf = sWi1[ntf * 16 + nl];
      float wg = sWi1[ntg * 16 + nl], wo = sWi1[nto * 16 + nl];
      v8f ai, af, ag, ao;
#pragma unroll
      for (int v = 0; v < 8; ++v) {
        ai[v] = fmaf(xm[v], wi, bi);
        af[v] = fmaf(xm[v], wf, bf);
        ag[v] = fmaf(xm[v], wg, bg);
        ao[v] = fmaf(xm[v], wo, b4);
      }
      ai = wmma_f16(h1A, load_B(sWhh1, nti, lane), ai);
      af = wmma_f16(h1A, load_B(sWhh1, ntf, lane), af);
      ag = wmma_f16(h1A, load_B(sWhh1, ntg, lane), ag);
      ao = wmma_f16(h1A, load_B(sWhh1, nto, lane), ao);
#pragma unroll
      for (int v = 0; v < 8; ++v) {
        float iv = fast_sigmoid(ai[v]);
        float fv = fast_sigmoid(af[v]);
        float gv = fast_tanh(ag[v]);
        float ov = fast_sigmoid(ao[v]);
        float cn = fv * c1r[ht][v] + iv * gv;
        c1r[ht][v] = cn;
        float hn = ov * fast_tanh(cn);
        sH1[(v + 8 * hl) * H1 + ht * 16 + nl] = (_Float16)hn;  // h1_t
      }
    }
    lds_fence();
    h1A       = load_A(sH1, H1, nl, hl);        // h1_t (reused next step too)
    v16h h2A0 = load_A(sH2, H2, nl, hl);        // h2_{t-1}, K = 0..31
    v16h h2A1 = load_A(sH2 + 32, H2, nl, hl);   // h2_{t-1}, K = 32..63

    // ------------- layer 2: gates = h1_t @ W_ih2^T + h2 @ W_hh2^T + b ------
    float pout[8];
#pragma unroll
    for (int v = 0; v < 8; ++v) pout[v] = 0.f;
#pragma unroll
    for (int ht = 0; ht < 4; ++ht) {
      const int nti = ht, ntf = 4 + ht, ntg = 8 + ht, nto = 12 + ht;
      float bi = sB2[nti * 16 + nl], bf = sB2[ntf * 16 + nl];
      float bg = sB2[ntg * 16 + nl], b4 = sB2[nto * 16 + nl];
      v8f ai, af, ag, ao;
#pragma unroll
      for (int v = 0; v < 8; ++v) { ai[v] = bi; af[v] = bf; ag[v] = bg; ao[v] = b4; }
      ai = wmma_f16(h1A,  load_B(sWih2, nti, lane),         ai);
      ai = wmma_f16(h2A0, load_B(sWhh2, nti * 2 + 0, lane), ai);
      ai = wmma_f16(h2A1, load_B(sWhh2, nti * 2 + 1, lane), ai);
      af = wmma_f16(h1A,  load_B(sWih2, ntf, lane),         af);
      af = wmma_f16(h2A0, load_B(sWhh2, ntf * 2 + 0, lane), af);
      af = wmma_f16(h2A1, load_B(sWhh2, ntf * 2 + 1, lane), af);
      ag = wmma_f16(h1A,  load_B(sWih2, ntg, lane),         ag);
      ag = wmma_f16(h2A0, load_B(sWhh2, ntg * 2 + 0, lane), ag);
      ag = wmma_f16(h2A1, load_B(sWhh2, ntg * 2 + 1, lane), ag);
      ao = wmma_f16(h1A,  load_B(sWih2, nto, lane),         ao);
      ao = wmma_f16(h2A0, load_B(sWhh2, nto * 2 + 0, lane), ao);
      ao = wmma_f16(h2A1, load_B(sWhh2, nto * 2 + 1, lane), ao);

      float wo2 = sWo[ht * 16 + nl];
#pragma unroll
      for (int v = 0; v < 8; ++v) {
        float iv = fast_sigmoid(ai[v]);
        float fv = fast_sigmoid(af[v]);
        float gv = fast_tanh(ag[v]);
        float ov = fast_sigmoid(ao[v]);
        float cn = fv * c2r[ht][v] + iv * gv;
        c2r[ht][v] = cn;
        float hn = ov * fast_tanh(cn);
        sH2[(v + 8 * hl) * H2 + ht * 16 + nl] = (_Float16)hn;  // h2_t
        pout[v] = fmaf(hn, wo2, pout[v]);  // partial of h2 . W_out
      }
    }
    lds_fence();

    // ------------- output: reduce over 16 columns within each half-wave ----
#pragma unroll
    for (int v = 0; v < 8; ++v) {
      float s = pout[v];
      s += __shfl_xor(s, 1, 32);
      s += __shfl_xor(s, 2, 32);
      s += __shfl_xor(s, 4, 32);
      s += __shfl_xor(s, 8, 32);
      pout[v] = s;
    }
    if (nl == 0) {  // lanes 0 and 16 each own 8 rows
#pragma unroll
      for (int v = 0; v < 8; ++v)
        out[(bbase + v + 8 * hl) * S + t] = pout[v] + bo;
    }
  }
}

extern "C" void kernel_launch(void* const* d_in, const int* in_sizes, int n_in,
                              void* d_out, int out_size, void* d_ws, size_t ws_size,
                              hipStream_t stream) {
  const float* inputs = (const float*)d_in[0];
  const float* h1     = (const float*)d_in[1];
  const float* c1     = (const float*)d_in[2];
  const float* h2     = (const float*)d_in[3];
  const float* c2     = (const float*)d_in[4];
  const float* W_ih1  = (const float*)d_in[5];
  const float* W_hh1  = (const float*)d_in[6];
  const float* b_ih1  = (const float*)d_in[7];
  const float* b_hh1  = (const float*)d_in[8];
  const float* W_ih2  = (const float*)d_in[9];
  const float* W_hh2  = (const float*)d_in[10];
  const float* b_ih2  = (const float*)d_in[11];
  const float* b_hh2  = (const float*)d_in[12];
  const float* W_out  = (const float*)d_in[13];
  const float* b_out  = (const float*)d_in[14];
  float* out = (float*)d_out;

  const int B = in_sizes[1] / H1;      // h1 is (B, 32)
  const int S = in_sizes[0] / B;       // inputs is (B, S)

  dim3 grid(B / 16), block(32);        // one wave32 per 16-row batch tile
  lstm_scan_kernel<<<grid, block, 0, stream>>>(
      inputs, h1, c1, h2, c2, W_ih1, W_hh1, b_ih1, b_hh1,
      W_ih2, W_hh2, b_ih2, b_hh2, W_out, b_out, out, S);
}
